// spikingAuxiliaryConvolutions_70909910057099
// MI455X (gfx1250) — matrix operand order repaired and checked by reading, not compiled, not run on hardware
//
#include <hip/hip_runtime.h>

typedef __attribute__((ext_vector_type(16))) __bf16 v16bf;
typedef __attribute__((ext_vector_type(8)))  float  v8f;
typedef __attribute__((ext_vector_type(4)))  int    v4i_;

#define SNN_DECAY  0.2f
#define SNN_THRESH 0.5f
#define SNN_EPS    1e-5f
#define NB 64          // batch

#if __has_builtin(__builtin_amdgcn_global_load_async_to_lds_b128) && \
    __has_builtin(__builtin_amdgcn_s_wait_asynccnt)
#define HAVE_ASYNC_LDS 1
typedef __attribute__((address_space(1))) v4i_ gv4i;
typedef __attribute__((address_space(3))) v4i_ sv4i;
#else
#define HAVE_ASYNC_LDS 0
#endif

// ---------------------------------------------------------------------------
// Fragment-major ("swizzled") global layouts, so WMMA fragments are contiguous
// 32-byte runs per lane.
//
// B matrix (K x N), B fragment, contiguous k order used here: lane = ((k>>4)&1)*16 + (n&15),
// element position p = k&15. Layout: [k/32][n/16][lane][p]
// ---------------------------------------------------------------------------
__device__ __host__ __forceinline__ size_t bswz(int k, int n, int N) {
  return ((size_t)((k >> 5) * (N >> 4) + (n >> 4)) * 512)
         + (size_t)(((k >> 4) & 1) * 256) + (size_t)((n & 15) * 16) + (size_t)(k & 15);
}
// A matrix (M x K), A fragment: lane = ((k>>4)&1)*16 + (m&15),
// position p = k&15. Layout: [m/16][k/32][lane][p]
__device__ __host__ __forceinline__ size_t aswz(int m, int k, int K) {
  int kl = k & 31;
  int p  = kl & 15;
  int h  = (kl >> 4) & 1;
  return ((size_t)((m >> 4) * (K >> 5) + (k >> 5)) * 512)
         + (size_t)(h * 256) + (size_t)((m & 15) * 16) + (size_t)p;
}

// ---------------------------------------------------------------------------
// weights f32 (row-major MxK, OIHW flat) -> bf16 fragment-major A layout
// ---------------------------------------------------------------------------
__global__ void k_convert_w(const float* __restrict__ in, __bf16* __restrict__ out,
                            int M, int K) {
  int i = blockIdx.x * blockDim.x + threadIdx.x;
  if (i >= M * K) return;
  int m = i / K, k = i % K;
  out[aswz(m, k, K)] = (__bf16)in[i];
}

// ---------------------------------------------------------------------------
// conv7_feats (N,C,19,19) f32 -> bf16 B layout with k=c, n = nb*361+p, N=23104
// ---------------------------------------------------------------------------
__global__ void k_transpose_x0(const float* __restrict__ in, __bf16* __restrict__ out) {
  const int total = 1024 * NB * 361;
  int i = blockIdx.x * blockDim.x + threadIdx.x;
  if (i >= total) return;
  int c = i / (NB * 361);
  int r = i % (NB * 361);
  int n = r / 361;
  int p = r % 361;
  out[bswz(c, r, NB * 361)] = (__bf16)in[(size_t)n * 1024 * 361 + (size_t)c * 361 + p];
}

// ---------------------------------------------------------------------------
// im2col 3x3: in [Cin][NB*IH*IW] row-major bf16 -> B layout [Cin*9 x NB*OH*OW]
// ---------------------------------------------------------------------------
__global__ void k_im2col3x3(const __bf16* __restrict__ in, __bf16* __restrict__ out,
                            int Cin, int IH, int IW, int OH, int OW, int stride, int pad) {
  int Ng = NB * OH * OW;
  int total = Cin * 9 * Ng;
  int i = blockIdx.x * blockDim.x + threadIdx.x;
  if (i >= total) return;
  int k = i / Ng, j = i % Ng;
  int c = k / 9, rr = k % 9, ky = rr / 3, kx = rr % 3;
  int n = j / (OH * OW);
  int p = j % (OH * OW);
  int oy = p / OW, ox = p % OW;
  int iy = oy * stride - pad + ky;
  int ix = ox * stride - pad + kx;
  __bf16 v = (__bf16)0.0f;
  if (iy >= 0 && iy < IH && ix >= 0 && ix < IW)
    v = in[(size_t)c * NB * IH * IW + (size_t)n * IH * IW + iy * IW + ix];
  out[bswz(k, j, Ng)] = v;
}

// ---------------------------------------------------------------------------
// WMMA bf16 GEMM: C[MxN] = A*B + bias, optional ReLU.
// A: fragment-major (weights, cache-resident, loaded straight from global).
// B: fragment-major, K-step tile (32x64 = 4KB) staged to LDS (async when
//    available), fragments read as contiguous 32B per lane.
// Block: 256 thr = 8 waves as 4(M) x 2(N); each wave -> 32x32 C via 2x2
// accumulators -> 4 v_wmma per K-step. Block tile 128(M) x 64(N) x 32(K).
// ---------------------------------------------------------------------------
__global__ __launch_bounds__(256)
void k_gemm_wmma_bf16(const __bf16* __restrict__ A, const __bf16* __restrict__ B,
                      const float* __restrict__ bias,
                      float* __restrict__ Cf, __bf16* __restrict__ Cb,
                      int M, int N, int K, int relu) {
  __shared__ __bf16 Bs[2048];          // [4 nblk][32 lane][16 p]

  const int t    = threadIdx.x;
  const int wave = t >> 5;
  const int lane = t & 31;
  const int half = lane >> 4;
  const int r    = lane & 15;
  const int wm   = wave & 3;           // 4 M-waves * 32 rows
  const int wn   = wave >> 2;          // 2 N-waves * 32 cols
  const int mBase = blockIdx.y * 128;
  const int nBase = blockIdx.x * 64;
  const int nKblk = K >> 5;

  v8f acc[2][2];
  #pragma unroll
  for (int a = 0; a < 2; ++a)
    #pragma unroll
    for (int b = 0; b < 2; ++b)
      acc[a][b] = (v8f){0.f, 0.f, 0.f, 0.f, 0.f, 0.f, 0.f, 0.f};

  const __bf16* Abase[2];
  #pragma unroll
  for (int ms = 0; ms < 2; ++ms) {
    int mblk = (mBase + wm * 32 + ms * 16) >> 4;
    Abase[ms] = A + (size_t)mblk * nKblk * 512 + lane * 16;
  }
  const __bf16*  Bg       = B + (size_t)(nBase >> 4) * 512;
  const size_t   bKstride = (size_t)(N >> 4) * 512;

  for (int kb = 0; kb < nKblk; ++kb) {
    // ---- stage B K-tile (4KB contiguous, 16B per thread) ----
    const __bf16* src = Bg + (size_t)kb * bKstride + t * 8;
#if HAVE_ASYNC_LDS
    __builtin_amdgcn_global_load_async_to_lds_b128(
        (gv4i*)(uintptr_t)src, (sv4i*)&Bs[t * 8], 0, 0);
    if (kb + 1 < nKblk)
      __builtin_prefetch(Bg + (size_t)(kb + 1) * bKstride + t * 8, 0, 0);
    __builtin_amdgcn_s_wait_asynccnt(0);
#else
    *(float4*)&Bs[t * 8] = *(const float4*)src;
    if (kb + 1 < nKblk)
      __builtin_prefetch(Bg + (size_t)(kb + 1) * bKstride + t * 8, 0, 0);
#endif
    __syncthreads();

    // ---- fragments: contiguous 32B per lane ----
    v16bf afrag[2], bfrag[2];
    #pragma unroll
    for (int ms = 0; ms < 2; ++ms)
      afrag[ms] = *(const v16bf*)(Abase[ms] + (size_t)kb * 512);
    #pragma unroll
    for (int ns = 0; ns < 2; ++ns)
      bfrag[ns] = *(const v16bf*)&Bs[(wn * 2 + ns) * 512 + lane * 16];

    #pragma unroll
    for (int ms = 0; ms < 2; ++ms)
      #pragma unroll
      for (int ns = 0; ns < 2; ++ns)
        acc[ms][ns] = __builtin_amdgcn_wmma_f32_16x16x32_bf16(
            false, afrag[ms], false, bfrag[ns], (short)0, acc[ms][ns], false, false);
    __syncthreads();
  }

  // ---- epilogue: D lane r = column; VGPR i = row (i + 8*half) ----
  #pragma unroll
  for (int ms = 0; ms < 2; ++ms) {
    #pragma unroll
    for (int ns = 0; ns < 2; ++ns) {
      const int n = nBase + wn * 32 + ns * 16 + r;
      #pragma unroll
      for (int i = 0; i < 8; ++i) {
        int m = mBase + wm * 32 + ms * 16 + half * 8 + i;
        float v = acc[ms][ns][i] + bias[m];
        if (relu) v = fmaxf(v, 0.f);
        if (Cf) Cf[(size_t)m * N + n] = v;
        if (Cb) Cb[(size_t)m * N + n] = (__bf16)v;
      }
    }
  }
}

// ---------------------------------------------------------------------------
// Fused LIF mem_update + InstanceNorm; one wave per (c,nb) group.
// spikes in {0,1} => var = mean - mean^2. bf16 copy written in B layout.
// ---------------------------------------------------------------------------
__global__ void k_memupdate_inorm(const float* __restrict__ conv,
                                  float* __restrict__ mem,
                                  float* __restrict__ spk,
                                  __bf16* __restrict__ spkbf,
                                  int HW, int groups) {
  int g = blockIdx.x * (blockDim.x >> 5) + (threadIdx.x >> 5);
  if (g >= groups) return;
  int lane = threadIdx.x & 31;
  size_t off = (size_t)g * HW;
  int c  = g / NB;
  int nb = g - c * NB;
  int Ngem = NB * HW;

  float s_local[4];
  float sum = 0.f;
  int cnt = 0;
  for (int e = lane; e < HW; e += 32) {
    float m = mem[off + e] * SNN_DECAY * (1.f - spk[off + e]) + conv[off + e];
    float s = (m - SNN_THRESH > 0.f) ? 1.f : 0.f;
    mem[off + e] = m;
    s_local[cnt++] = s;
    sum += s;
  }
  #pragma unroll
  for (int o = 16; o > 0; o >>= 1) sum += __shfl_xor(sum, o, 32);
  float mean = sum / (float)HW;
  float var  = mean - mean * mean;
  float inv  = rsqrtf(var + SNN_EPS);
  cnt = 0;
  for (int e = lane; e < HW; e += 32) {
    float o = (s_local[cnt++] - mean) * inv;
    spk[off + e] = o;
    if (spkbf) spkbf[bswz(c, nb * HW + e, Ngem)] = (__bf16)o;
  }
}

// plain mem_update (a4: HW==1, no InstanceNorm, feeds nothing downstream)
__global__ void k_memupdate(const float* __restrict__ conv, float* __restrict__ mem,
                            float* __restrict__ spk, int n) {
  int i = blockIdx.x * blockDim.x + threadIdx.x;
  if (i >= n) return;
  float m = mem[i] * SNN_DECAY * (1.f - spk[i]) + conv[i];
  float s = (m - SNN_THRESH > 0.f) ? 1.f : 0.f;
  mem[i] = m;
  spk[i] = s;
}

// [C][NB*HW] -> NCHW float output
__global__ void k_copy_out(const float* __restrict__ in, float* __restrict__ out,
                           int C, int HW) {
  int total = NB * C * HW;
  int i = blockIdx.x * blockDim.x + threadIdx.x;
  if (i >= total) return;
  int n  = i / (C * HW);
  int rr = i % (C * HW);
  int c  = rr / HW;
  int hw = rr % HW;
  out[i] = in[(size_t)c * NB * HW + (size_t)n * HW + hw];
}

// ---------------------------------------------------------------------------
// host side
// ---------------------------------------------------------------------------
static inline int ceil_div(int a, int b) { return (a + b - 1) / b; }

extern "C" void kernel_launch(void* const* d_in, const int* in_sizes, int n_in,
                              void* d_out, int out_size, void* d_ws, size_t ws_size,
                              hipStream_t stream) {
  (void)in_sizes; (void)n_in; (void)out_size; (void)ws_size;
  const float* conv7 = (const float*)d_in[0];
  const float* w_f32[8] = {(const float*)d_in[1],  (const float*)d_in[3],
                           (const float*)d_in[5],  (const float*)d_in[7],
                           (const float*)d_in[9],  (const float*)d_in[11],
                           (const float*)d_in[13], (const float*)d_in[15]};
  const float* bias[8]  = {(const float*)d_in[2],  (const float*)d_in[4],
                           (const float*)d_in[6],  (const float*)d_in[8],
                           (const float*)d_in[10], (const float*)d_in[12],
                           (const float*)d_in[14], (const float*)d_in[16]};
  const int wM[8] = {256, 512, 128, 256, 128, 256, 128, 256};
  const int wK[8] = {1024, 2304, 512, 1152, 256, 1152, 256, 1152};

  char* ws = (char*)d_ws;
  size_t off = 0;
  auto alloc = [&](size_t bytes) -> char* {
    char* p = ws + off;
    off = (off + bytes + 255) & ~(size_t)255;
    return p;
  };

  __bf16* X0   = (__bf16*)alloc((size_t)1024 * NB * 361 * 2);
  __bf16* Y1   = (__bf16*)alloc((size_t)256 * NB * 361 * 2);
  __bf16* B2   = (__bf16*)alloc((size_t)2304 * NB * 100 * 2);
  float*  Cbuf = (float*) alloc((size_t)512 * NB * 100 * 4);
  float*  a1m  = (float*) alloc((size_t)512 * NB * 100 * 4);
  float*  a1s  = (float*) alloc((size_t)512 * NB * 100 * 4);
  __bf16* a1sb = (__bf16*)alloc((size_t)512 * NB * 100 * 2);
  __bf16* Y3   = (__bf16*)alloc((size_t)128 * NB * 100 * 2);
  __bf16* B4   = (__bf16*)alloc((size_t)1152 * NB * 25 * 2);
  float*  a2m  = (float*) alloc((size_t)256 * NB * 25 * 4);
  float*  a2s  = (float*) alloc((size_t)256 * NB * 25 * 4);
  __bf16* a2sb = (__bf16*)alloc((size_t)256 * NB * 25 * 2);
  __bf16* Y5   = (__bf16*)alloc((size_t)128 * NB * 25 * 2);
  __bf16* B6   = (__bf16*)alloc((size_t)1152 * NB * 9 * 2);
  float*  a3m  = (float*) alloc((size_t)256 * NB * 9 * 4);
  float*  a3s  = (float*) alloc((size_t)256 * NB * 9 * 4);
  __bf16* a3sb = (__bf16*)alloc((size_t)256 * NB * 9 * 2);
  __bf16* Y7   = (__bf16*)alloc((size_t)128 * NB * 9 * 2);
  __bf16* B8   = (__bf16*)alloc((size_t)1152 * NB * 2);
  float*  a4m  = (float*) alloc((size_t)256 * NB * 4);
  float*  a4s  = (float*) alloc((size_t)256 * NB * 4);
  __bf16* Wbf[8];
  for (int l = 0; l < 8; ++l) Wbf[l] = (__bf16*)alloc((size_t)wM[l] * wK[l] * 2);

  (void)hipMemsetAsync(a1m, 0, (size_t)512 * NB * 100 * 4, stream);
  (void)hipMemsetAsync(a1s, 0, (size_t)512 * NB * 100 * 4, stream);
  (void)hipMemsetAsync(a2m, 0, (size_t)256 * NB * 25 * 4, stream);
  (void)hipMemsetAsync(a2s, 0, (size_t)256 * NB * 25 * 4, stream);
  (void)hipMemsetAsync(a3m, 0, (size_t)256 * NB * 9 * 4, stream);
  (void)hipMemsetAsync(a3s, 0, (size_t)256 * NB * 9 * 4, stream);
  (void)hipMemsetAsync(a4m, 0, (size_t)256 * NB * 4, stream);
  (void)hipMemsetAsync(a4s, 0, (size_t)256 * NB * 4, stream);

  for (int l = 0; l < 8; ++l) {
    int n = wM[l] * wK[l];
    k_convert_w<<<ceil_div(n, 256), 256, 0, stream>>>(w_f32[l], Wbf[l], wM[l], wK[l]);
  }
  {
    int n = 1024 * NB * 361;
    k_transpose_x0<<<ceil_div(n, 256), 256, 0, stream>>>(conv7, X0);
  }

  auto gemm = [&](const __bf16* A, const __bf16* B, const float* b,
                  float* Cf, __bf16* Cb, int M, int N, int K, int relu) {
    dim3 grid(N / 64, M / 128);
    k_gemm_wmma_bf16<<<grid, 256, 0, stream>>>(A, B, b, Cf, Cb, M, N, K, relu);
  };
  auto im2col = [&](const __bf16* in, __bf16* out, int Cin, int IH, int IW,
                    int OH, int OW, int stride, int pad) {
    int n = Cin * 9 * NB * OH * OW;
    k_im2col3x3<<<ceil_div(n, 256), 256, 0, stream>>>(in, out, Cin, IH, IW, OH, OW, stride, pad);
  };

  for (int tstep = 0; tstep < 8; ++tstep) {
    gemm(Wbf[0], X0, bias[0], nullptr, Y1, 256, NB * 361, 1024, 1);        // conv8_1
    im2col(Y1, B2, 256, 19, 19, 10, 10, 2, 1);                              // 19->10
    gemm(Wbf[1], B2, bias[1], Cbuf, nullptr, 512, NB * 100, 2304, 0);      // conv8_2
    k_memupdate_inorm<<<ceil_div(512 * NB, 8), 256, 0, stream>>>(Cbuf, a1m, a1s, a1sb, 100, 512 * NB);
    gemm(Wbf[2], a1sb, bias[2], nullptr, Y3, 128, NB * 100, 512, 1);       // conv9_1
    im2col(Y3, B4, 128, 10, 10, 5, 5, 2, 1);                                // 10->5
    gemm(Wbf[3], B4, bias[3], Cbuf, nullptr, 256, NB * 25, 1152, 0);       // conv9_2
    k_memupdate_inorm<<<ceil_div(256 * NB, 8), 256, 0, stream>>>(Cbuf, a2m, a2s, a2sb, 25, 256 * NB);
    gemm(Wbf[4], a2sb, bias[4], nullptr, Y5, 128, NB * 25, 256, 1);        // conv10_1
    im2col(Y5, B6, 128, 5, 5, 3, 3, 1, 0);                                  // 5->3
    gemm(Wbf[5], B6, bias[5], Cbuf, nullptr, 256, NB * 9, 1152, 0);        // conv10_2
    k_memupdate_inorm<<<ceil_div(256 * NB, 8), 256, 0, stream>>>(Cbuf, a3m, a3s, a3sb, 9, 256 * NB);
    gemm(Wbf[6], a3sb, bias[6], nullptr, Y7, 128, NB * 9, 256, 1);         // conv11_1
    im2col(Y7, B8, 128, 3, 3, 1, 1, 1, 0);                                  // 3->1
    gemm(Wbf[7], B8, bias[7], Cbuf, nullptr, 256, NB, 1152, 0);            // conv11_2
    k_memupdate<<<ceil_div(256 * NB, 256), 256, 0, stream>>>(Cbuf, a4m, a4s, 256 * NB);
  }

  float* out = (float*)d_out;
  k_copy_out<<<ceil_div(NB * 512 * 100, 256), 256, 0, stream>>>(a1s, out, 512, 100);
  k_copy_out<<<ceil_div(NB * 256 * 25, 256), 256, 0, stream>>>(a2s, out + 3276800, 256, 25);
  k_copy_out<<<ceil_div(NB * 256 * 9, 256), 256, 0, stream>>>(a3s, out + 3686400, 256, 9);
  k_copy_out<<<ceil_div(NB * 256 * 1, 256), 256, 0, stream>>>(a4s, out + 3833856, 256, 1);
}
